// Recommender_38654705664771
// MI455X (gfx1250) — matrix operand order repaired and checked by reading, not compiled, not run on hardware
//
#include <hip/hip_runtime.h>
#include <hip/hip_bf16.h>

typedef __attribute__((ext_vector_type(16))) __bf16 v16bf;
typedef __attribute__((ext_vector_type(8)))  float  v8f;

#define D   256
#define KT  8                      // 256 / 32 k-steps per WMMA chain
#define WPACK_ELEMS (D * D)        // 65536 bf16 per packed weight matrix
#define HS  272                    // bf16 LDS row stride (16B aligned rows)
#define FS  264                    // f32 LDS staging row stride
#define STEP_SMEM (3 * 16 * HS * 2 + 5 * 16 * FS * 4)   // 110592 bytes

enum { WZ = 0, UZ, WR, UR, WH, UH, AH, AUm, WT, WG };   // packed weight order

union BF16V { v16bf v; __bf16 e[16]; };
union F8V   { v8f   v; float  e[8];  };
union F16R  { float4 q4[4]; float e[16]; };

__device__ __forceinline__ v8f wmma_bf16(v16bf a, v16bf b, v8f c) {
  return __builtin_amdgcn_wmma_f32_16x16x32_bf16(false, a, false, b, (short)0, c,
                                                 false, false);
}

// B fragment: packed so each lane reads 32 contiguous bytes per (ntile,ktile).
__device__ __forceinline__ v16bf load_bfrag(const __bf16* Wp, int nt, int kt, int lane) {
  const __bf16* q = Wp + (((((nt << 3) + kt) << 5) + lane) << 4);
  BF16V f;
  ((int4*)f.e)[0] = ((const int4*)q)[0];
  ((int4*)f.e)[1] = ((const int4*)q)[1];
  return f.v;
}

// A fragment from row-major bf16 in LDS (stride HS). Lane l<16: row l, K {0..7,16..23};
// lane>=16: row l-16, K {8..15,24..31}  (CDNA5 16-bit A layout).
__device__ __forceinline__ v16bf load_afrag_lds(const __bf16* base, int kt, int lane) {
  const int r = lane & 15, hs = (lane >> 4) & 1;
  const __bf16* q = base + r * HS + (kt << 5) + (hs << 3);
  BF16V f;
  ((int4*)f.e)[0] = *(const int4*)q;
  ((int4*)f.e)[1] = *(const int4*)(q + 16);
  return f.v;
}

// Raw f32 A-row chunk load (converted to bf16 later so cvt overlaps the loads).
__device__ __forceinline__ void load_araw(F16R& t, const float* row, int kt, int hs) {
  const float* q = row + (kt << 5) + (hs << 3);
  t.q4[0] = ((const float4*)q)[0];
  t.q4[1] = ((const float4*)q)[1];
  t.q4[2] = ((const float4*)(q + 16))[0];
  t.q4[3] = ((const float4*)(q + 16))[1];
}

__device__ __forceinline__ v16bf cvt_afrag(const F16R& t) {
  BF16V f;
#pragma unroll
  for (int i = 0; i < 16; i++) f.e[i] = (__bf16)t.e[i];
  return f.v;
}

// ---------------------------------------------------------------- weight pack
__global__ void cara_pack(const float* __restrict__ W, __bf16* __restrict__ out) {
  const int i = blockIdx.x * 256 + threadIdx.x;      // 0..65535
  const int e = i & 15, lane = (i >> 4) & 31, kt = (i >> 9) & 7, nt = i >> 12;
  const int k = (kt << 5) + (e & 7) + ((e >> 3) << 4) + ((lane < 16) ? 0 : 8);   // B layout K mapping
  const int n = (nt << 4) + (lane & 15);
  out[i] = (__bf16)W[k * D + n];
}

// ------------------------------------------- batched GEMM with row gather
// C[row,:] = gather(Emb, idx)[row,:] @ W (+b0 +b1), bf16 out. Wave = 16 rows x 256 cols.
// Fully unrolled: depth-4 B-fragment software pipeline + one-kt-ahead A prefetch.
__global__ __launch_bounds__(256) void cara_gemm(
    const float* __restrict__ Emb, const int* __restrict__ idx,
    const __bf16* __restrict__ Wp, const float* __restrict__ b0,
    const float* __restrict__ b1, __bf16* __restrict__ out) {
  const int lane = threadIdx.x & 31, wave = threadIdx.x >> 5;
  const int row0 = (blockIdx.x * 8 + wave) * 16;
  const int r = lane & 15, hs = (lane >> 4) & 1;
  const int row = row0 + r;
  const float* arow = Emb + (size_t)(idx ? idx[row] : row) * D;

  v8f acc[16] = {};
  F16R araw;
  load_araw(araw, arow, 0, hs);
  v16bf bbuf[4];
#pragma unroll
  for (int p = 0; p < 4; p++) bbuf[p] = load_bfrag(Wp, p, 0, lane);

#pragma unroll
  for (int kt = 0; kt < KT; kt++) {
    const v16bf af = cvt_afrag(araw);
    if (kt < KT - 1) load_araw(araw, arow, kt + 1, hs);
#pragma unroll
    for (int nt = 0; nt < 12; nt++) {
      acc[nt] = wmma_bf16(af, bbuf[nt & 3], acc[nt]);
      bbuf[nt & 3] = load_bfrag(Wp, nt + 4, kt, lane);
    }
#pragma unroll
    for (int nt = 12; nt < 16; nt++) {
      acc[nt] = wmma_bf16(af, bbuf[nt & 3], acc[nt]);
      if (kt < KT - 1) bbuf[nt & 3] = load_bfrag(Wp, nt - 12, kt + 1, lane);
    }
  }

#pragma unroll
  for (int nt = 0; nt < 16; nt++) {
    F8V a; a.v = acc[nt];
    const int c = nt * 16 + r;
    const float bias = (b0 ? b0[c] : 0.f) + (b1 ? b1[c] : 0.f);
#pragma unroll
    for (int v = 0; v < 8; v++)
      out[(size_t)(row0 + hs * 8 + v) * D + c] = (__bf16)(a.e[v] + bias);
  }
}

// ------------------- fused t1*g1 kernel: two GEMMs sharing A + tanh epilogue
__global__ __launch_bounds__(256) void cara_tg(
    const float* __restrict__ Emb, const int* __restrict__ idx,
    const __bf16* __restrict__ WpT, const __bf16* __restrict__ WpG,
    const float* __restrict__ Ut, const float* __restrict__ Ug,
    const float* __restrict__ bt, const float* __restrict__ bg,
    const float* __restrict__ gap, const float* __restrict__ dist,
    __bf16* __restrict__ out) {
  const int lane = threadIdx.x & 31, wave = threadIdx.x >> 5;
  const int row0 = (blockIdx.x * 4 + (wave >> 1)) * 16;
  const int nh8 = (wave & 1) * 8;
  const int r = lane & 15, hs = (lane >> 4) & 1;
  const int row = row0 + r;
  const float* arow = Emb + (size_t)idx[row] * D;

  v8f at[8] = {}, ag[8] = {};
  F16R araw;
  load_araw(araw, arow, 0, hs);
  v16bf bT[2], bG[2];
  bT[0] = load_bfrag(WpT, nh8 + 0, 0, lane);
  bG[0] = load_bfrag(WpG, nh8 + 0, 0, lane);
  bT[1] = load_bfrag(WpT, nh8 + 1, 0, lane);
  bG[1] = load_bfrag(WpG, nh8 + 1, 0, lane);

#pragma unroll
  for (int kt = 0; kt < KT; kt++) {
    const v16bf af = cvt_afrag(araw);
    if (kt < KT - 1) load_araw(araw, arow, kt + 1, hs);
#pragma unroll
    for (int j = 0; j < 6; j++) {
      at[j] = wmma_bf16(af, bT[j & 1], at[j]);
      ag[j] = wmma_bf16(af, bG[j & 1], ag[j]);
      bT[j & 1] = load_bfrag(WpT, nh8 + j + 2, kt, lane);
      bG[j & 1] = load_bfrag(WpG, nh8 + j + 2, kt, lane);
    }
#pragma unroll
    for (int j = 6; j < 8; j++) {
      at[j] = wmma_bf16(af, bT[j & 1], at[j]);
      ag[j] = wmma_bf16(af, bG[j & 1], ag[j]);
      if (kt < KT - 1) {
        bT[j & 1] = load_bfrag(WpT, nh8 + j - 6, kt + 1, lane);
        bG[j & 1] = load_bfrag(WpG, nh8 + j - 6, kt + 1, lane);
      }
    }
  }

#pragma unroll
  for (int j = 0; j < 8; j++) {
    F8V t_, g_; t_.v = at[j]; g_.v = ag[j];
    const int c = (nh8 + j) * 16 + r;
#pragma unroll
    for (int v = 0; v < 8; v++) {
      const int rr = row0 + hs * 8 + v;
      const float t1 = tanhf(t_.e[v] + tanhf(gap[rr]  * Ut[c]) + bt[c]);
      const float g1 = tanhf(g_.e[v] + tanhf(dist[rr] * Ug[c]) + bg[c]);
      out[(size_t)rr * D + c] = (__bf16)(t1 * g1);
    }
  }
}

// ------------------------------- per-step in-workgroup GEMM stages (A in LDS)
// 3 matrices x 16 ntiles over 8 waves: 6 jobs per wave, depth-3 B pipeline.
__device__ __forceinline__ void mm3(const __bf16* A, const __bf16* W0,
                                    const __bf16* W1, const __bf16* W2,
                                    float* S0, float* S1, float* S2,
                                    int lane, int wave) {
  const __bf16* Wi[6];
  int nti[6];
#pragma unroll
  for (int i = 0; i < 6; i++) {
    const int job = wave + (i << 3);
    const int mat = job >> 4;
    Wi[i]  = (mat == 0) ? W0 : ((mat == 1) ? W1 : W2);
    nti[i] = job & 15;
  }
  v8f acc[6] = {};
  v16bf bbuf[3];
#pragma unroll
  for (int p = 0; p < 3; p++) bbuf[p] = load_bfrag(Wi[p], nti[p], 0, lane);

#pragma unroll
  for (int kt = 0; kt < KT; kt++) {
    const v16bf af = load_afrag_lds(A, kt, lane);
#pragma unroll
    for (int i = 0; i < 3; i++) {
      acc[i] = wmma_bf16(af, bbuf[i], acc[i]);
      bbuf[i] = load_bfrag(Wi[i + 3], nti[i + 3], kt, lane);
    }
#pragma unroll
    for (int i = 3; i < 6; i++) {
      acc[i] = wmma_bf16(af, bbuf[i - 3], acc[i]);
      if (kt < KT - 1) bbuf[i - 3] = load_bfrag(Wi[i - 3], nti[i - 3], kt + 1, lane);
    }
  }

  const int r = lane & 15, hs = (lane >> 4) & 1;
#pragma unroll
  for (int i = 0; i < 6; i++) {
    const int job = wave + (i << 3);
    const int mat = job >> 4, nt = job & 15;
    float* S = (mat == 0) ? S0 : ((mat == 1) ? S1 : S2);
    F8V a; a.v = acc[i];
#pragma unroll
    for (int v = 0; v < 8; v++)
      S[(hs * 8 + v) * FS + nt * 16 + r] = a.e[v];
  }
}

__device__ __forceinline__ void mm1(const __bf16* A, const __bf16* W, float* S,
                                    int lane, int wave) {
  v8f a0 = {}, a1 = {};
  v16bf b0 = load_bfrag(W, wave,     0, lane);
  v16bf b1 = load_bfrag(W, wave + 8, 0, lane);
#pragma unroll
  for (int kt = 0; kt < KT; kt++) {
    const v16bf af = load_afrag_lds(A, kt, lane);
    a0 = wmma_bf16(af, b0, a0);
    if (kt < KT - 1) b0 = load_bfrag(W, wave, kt + 1, lane);
    a1 = wmma_bf16(af, b1, a1);
    if (kt < KT - 1) b1 = load_bfrag(W, wave + 8, kt + 1, lane);
  }
  const int r = lane & 15, hs = (lane >> 4) & 1;
  F8V x0, x1; x0.v = a0; x1.v = a1;
#pragma unroll
  for (int v = 0; v < 8; v++) {
    S[(hs * 8 + v) * FS + wave * 16 + r]       = x0.e[v];
    S[(hs * 8 + v) * FS + (wave + 8) * 16 + r] = x1.e[v];
  }
}

// ------------------------------------------------------ fused recurrence step
struct StepParams {
  const __bf16 *XZ0, *XZ1, *XR0, *XR1, *XH0, *XH1, *TG0, *TG1;
  const __bf16 *AU, *UWZ, *UWR, *UWH;
  float *h0, *h1;
  const int* time_idx;
  const float* Temb;
  const float *bz, *br, *bh;
  const __bf16* Wp;
  int t, T;
};

__global__ __launch_bounds__(256) void cara_step(StepParams P) {
  extern __shared__ char smem[];
  __bf16* hbf  = (__bf16*)smem;                 // 16 x HS
  __bf16* ubbf = hbf  + 16 * HS;
  __bf16* rhbf = ubbf + 16 * HS;
  float*  S0 = (float*)(smem + 3 * 16 * HS * 2);
  float*  S1 = S0 + 16 * FS;
  float*  S2 = S1 + 16 * FS;
  float*  S3 = S2 + 16 * FS;
  float*  S4 = S3 + 16 * FS;

  const int seq = blockIdx.y;
  const __bf16* XZ = seq ? P.XZ1 : P.XZ0;
  const __bf16* XR = seq ? P.XR1 : P.XR0;
  const __bf16* XH = seq ? P.XH1 : P.XH0;
  const __bf16* TG = seq ? P.TG1 : P.TG0;
  float* h = seq ? P.h1 : P.h0;

  const int tid = threadIdx.x;
  const int lane = tid & 31, wave = tid >> 5;
  const int m0 = blockIdx.x * 16;
  const int orow = tid >> 4;              // owned row (0..15)
  const int oc0 = (tid & 15) << 4;        // owned 16-col chunk
  const int b = m0 + orow;
  const size_t g = ((size_t)b * P.T + P.t) * D + oc0;
  float* hrow = h + (size_t)b * D + oc0;

  float hreg[16];
#pragma unroll
  for (int i = 0; i < 16; i++) {
    hreg[i] = hrow[i];
    hbf[orow * HS + oc0 + i] = (__bf16)hreg[i];
  }
  __syncthreads();

  // stage 1:  S0 = h@A_h,  S1 = h@U_z,  S2 = h@U_r
  mm3(hbf, P.Wp + AH * WPACK_ELEMS, P.Wp + UZ * WPACK_ELEMS,
      P.Wp + UR * WPACK_ELEMS, S0, S1, S2, lane, wave);
  __syncthreads();

  {  // a = tanh(h@A_h + u@A_u + b);  ub = a*u
    const int ti = P.time_idx[b * P.T + P.t];
    const float* urow = P.Temb + (size_t)ti * D + oc0;
#pragma unroll
    for (int i = 0; i < 16; i++) {
      const float a = tanhf(S0[orow * FS + oc0 + i] + (float)P.AU[g + i]);
      ubbf[orow * HS + oc0 + i] = (__bf16)(a * urow[i]);
    }
  }
  __syncthreads();

  // stage 2:  S0 = ub@W_z,  S3 = ub@W_r,  S4 = ub@W_h
  mm3(ubbf, P.Wp + WZ * WPACK_ELEMS, P.Wp + WR * WPACK_ELEMS,
      P.Wp + WH * WPACK_ELEMS, S0, S3, S4, lane, wave);
  __syncthreads();

  float zreg[16], usum[16], xhq[16];
#pragma unroll
  for (int i = 0; i < 16; i++) {
    const int c = oc0 + i;
    const float bzv = P.bz[c], brv = P.br[c], bhv = P.bh[c];
    const float qz = S0[orow * FS + c], qr = S3[orow * FS + c], qh = S4[orow * FS + c];
    const float z = tanhf((float)XZ[g + i] + S1[orow * FS + c] + qz + bzv);
    const float r = tanhf((float)XR[g + i] + S2[orow * FS + c] + qr + brv);
    rhbf[orow * HS + c] = (__bf16)(r * (float)TG[g + i] * hreg[i]);
    usum[i] = ((float)P.UWZ[g + i] - qz + bzv) + ((float)P.UWR[g + i] - qr + brv) +
              ((float)P.UWH[g + i] - qh + bhv);
    xhq[i] = (float)XH[g + i] + qh + bhv;
    zreg[i] = z;
  }
  __syncthreads();

  // stage 3:  S1 = (r*t1*g1*h) @ U_h
  mm1(rhbf, P.Wp + UH * WPACK_ELEMS, S1, lane, wave);
  __syncthreads();

#pragma unroll
  for (int i = 0; i < 16; i++) {
    float v = xhq[i] + S1[orow * FS + oc0 + i];
    v = fminf(fmaxf(v * (1.0f / 6.0f) + 0.5f, 0.f), 1.f);          // hard_sigmoid
    const float hn = zreg[i] * hreg[i] + (1.f - zreg[i]) * v;
    hrow[i] = (1.f + usum[i]) * hn;
  }
}

// ---------------------------------------------------------------- tail kernels
__global__ void cara_zero(float* p, int n) {
  const int i = blockIdx.x * 256 + threadIdx.x;
  if (i < n) p[i] = 0.f;
}

__global__ void cara_dot(const float* __restrict__ hp, const float* __restrict__ hn,
                         const float* __restrict__ Uemb, const int* __restrict__ user,
                         float* __restrict__ partial) {
  __shared__ float red[256];
  const int b = blockIdx.x, c = threadIdx.x;
  const float ul = Uemb[(size_t)user[b] * D + c];
  red[c] = (hp[(size_t)b * D + c] - hn[(size_t)b * D + c]) * ul;
  __syncthreads();
  for (int s = 128; s > 0; s >>= 1) {
    if (c < s) red[c] += red[c + s];
    __syncthreads();
  }
  if (c == 0) partial[b] = red[0];
}

__global__ void cara_finalize(const float* __restrict__ partial, int B,
                              float* __restrict__ out, int n) {
  __shared__ float red[256];
  const int tid = threadIdx.x;
  float s = 0.f;
  for (int b = tid; b < B; b += 256) s += partial[b];
  red[tid] = s;
  __syncthreads();
  for (int st = 128; st > 0; st >>= 1) {
    if (tid < st) red[tid] += red[tid + st];
    __syncthreads();
  }
  const float diff = red[0];
  const float x = -diff;                                  // 1 - log(sigmoid(diff))
  const float sp = fmaxf(x, 0.f) + log1pf(expf(-fabsf(x)));
  const float res = 1.f + sp;
  for (int i = tid; i < n; i += 256) out[i] = res;
}

// ----------------------------------------------------------------------- host
extern "C" void kernel_launch(void* const* d_in, const int* in_sizes, int n_in,
                              void* d_out, int out_size, void* d_ws, size_t ws_size,
                              hipStream_t stream) {
  const int B = in_sizes[0];
  const int T = in_sizes[1] / B;
  const size_t BT = (size_t)B * T;

  const int*   user  = (const int*)d_in[0];
  const int*   timei = (const int*)d_in[1];
  const float* gap   = (const float*)d_in[2];
  const float* pdist = (const float*)d_in[3];
  const float* ndist = (const float*)d_in[4];
  const int*   chk   = (const int*)d_in[5];
  const int*   nchk  = (const int*)d_in[6];
  const float* Uemb  = (const float*)d_in[7];
  const float* Vemb  = (const float*)d_in[8];
  const float* Temb  = (const float*)d_in[9];
  const float* W[10];
  for (int i = 0; i < 10; i++) W[i] = (const float*)d_in[10 + i];  // WZ..WG order
  const float* Ut  = (const float*)d_in[20];
  const float* Ug  = (const float*)d_in[21];
  const float* bz  = (const float*)d_in[22];
  const float* br  = (const float*)d_in[23];
  const float* bh  = (const float*)d_in[24];
  const float* bah = (const float*)d_in[25];
  const float* bau = (const float*)d_in[26];
  const float* bt  = (const float*)d_in[27];
  const float* bg  = (const float*)d_in[28];

  char* wsp = (char*)d_ws;
  auto carve = [&](size_t bytes) -> char* {
    char* p = wsp; wsp += (bytes + 255) & ~(size_t)255; return p;
  };
  __bf16* packed = (__bf16*)carve((size_t)10 * WPACK_ELEMS * 2);
  auto bfarr = [&]() { return (__bf16*)carve(BT * D * 2); };
  __bf16 *XZp = bfarr(), *XRp = bfarr(), *XHp = bfarr();
  __bf16 *XZn = bfarr(), *XRn = bfarr(), *XHn = bfarr();
  __bf16 *TGp = bfarr(), *TGn = bfarr();
  __bf16 *AUa = bfarr(), *UWZa = bfarr(), *UWRa = bfarr(), *UWHa = bfarr();
  float* hbuf = (float*)carve((size_t)2 * B * D * 4);
  float* partial = (float*)carve((size_t)B * 4);
  float* hp = hbuf;
  float* hn = hbuf + (size_t)B * D;

  for (int m = 0; m < 10; m++)
    cara_pack<<<WPACK_ELEMS / 256, 256, 0, stream>>>(W[m], packed + (size_t)m * WPACK_ELEMS);

  const int gB = (int)(BT / 128);
  const __bf16* Pw = packed;
  cara_gemm<<<gB, 256, 0, stream>>>(Vemb, chk,  Pw + WZ  * WPACK_ELEMS, bz,  nullptr, XZp);
  cara_gemm<<<gB, 256, 0, stream>>>(Vemb, chk,  Pw + WR  * WPACK_ELEMS, br,  nullptr, XRp);
  cara_gemm<<<gB, 256, 0, stream>>>(Vemb, chk,  Pw + WH  * WPACK_ELEMS, bh,  nullptr, XHp);
  cara_gemm<<<gB, 256, 0, stream>>>(Vemb, nchk, Pw + WZ  * WPACK_ELEMS, bz,  nullptr, XZn);
  cara_gemm<<<gB, 256, 0, stream>>>(Vemb, nchk, Pw + WR  * WPACK_ELEMS, br,  nullptr, XRn);
  cara_gemm<<<gB, 256, 0, stream>>>(Vemb, nchk, Pw + WH  * WPACK_ELEMS, bh,  nullptr, XHn);
  cara_gemm<<<gB, 256, 0, stream>>>(Temb, timei, Pw + AUm * WPACK_ELEMS, bah, bau,    AUa);
  cara_gemm<<<gB, 256, 0, stream>>>(Temb, timei, Pw + WZ  * WPACK_ELEMS, nullptr, nullptr, UWZa);
  cara_gemm<<<gB, 256, 0, stream>>>(Temb, timei, Pw + WR  * WPACK_ELEMS, nullptr, nullptr, UWRa);
  cara_gemm<<<gB, 256, 0, stream>>>(Temb, timei, Pw + WH  * WPACK_ELEMS, nullptr, nullptr, UWHa);

  const int tgB = (int)(BT / 64);
  cara_tg<<<tgB, 256, 0, stream>>>(Vemb, chk,  Pw + WT * WPACK_ELEMS, Pw + WG * WPACK_ELEMS,
                                   Ut, Ug, bt, bg, gap, pdist, TGp);
  cara_tg<<<tgB, 256, 0, stream>>>(Vemb, nchk, Pw + WT * WPACK_ELEMS, Pw + WG * WPACK_ELEMS,
                                   Ut, Ug, bt, bg, gap, ndist, TGn);

  const int nz = 2 * B * D;
  cara_zero<<<(nz + 255) / 256, 256, 0, stream>>>(hbuf, nz);

  StepParams P;
  P.XZ0 = XZp; P.XZ1 = XZn; P.XR0 = XRp; P.XR1 = XRn;
  P.XH0 = XHp; P.XH1 = XHn; P.TG0 = TGp; P.TG1 = TGn;
  P.AU = AUa; P.UWZ = UWZa; P.UWR = UWRa; P.UWH = UWHa;
  P.h0 = hp; P.h1 = hn;
  P.time_idx = timei; P.Temb = Temb;
  P.bz = bz; P.br = br; P.bh = bh;
  P.Wp = packed; P.T = T;
  for (int t = 0; t < T; t++) {
    P.t = t;
    cara_step<<<dim3(B / 16, 2), 256, STEP_SMEM, stream>>>(P);
  }

  cara_dot<<<B, 256, 0, stream>>>(hp, hn, Uemb, user, partial);
  cara_finalize<<<1, 256, 0, stream>>>(partial, B, (float*)d_out, out_size);
}
